// LpAlignEntropyLoss_22076131901968
// MI455X (gfx1250) — matrix operand order repaired and checked by reading, not compiled, not run on hardware
//
#include <hip/hip_runtime.h>
#include <hip/hip_bf16.h>
#include <math.h>

// LpAlignEntropyLoss for MI455X (gfx1250).
// align   = mean ||v0_i - v1_i||_2
// entropy = avg over views of mean_i [ log(sum_{j!=i} exp(-d_ij)) - log(N-1) ]
// d_ij from Gram = Z Z^T computed with bf16x3 (hi*hi + hi*lo + lo*hi) WMMA,
// fp32 accumulate -> ~fp32 accuracy. All operand data (16 MB) is L2-resident
// (192 MB L2), so the kernel is compute-bound on v_wmma_f32_16x16x32_bf16.
// Staging uses CDNA5 GLOBAL_LOAD_ASYNC_TO_LDS_B128 (ASYNCcnt).
// k-loop unroll capped at 2 to keep VGPR pressure < 256 (avoid s_set_vgpr_msb
// traffic in the WMMA inner loop).

#define N_ROWS 8192
#define DIM    256
#define EPS_F  1e-12f

#define BM 64
#define BN 128
#define LDS_STRIDE 264   // 256 + 8 bf16 pad -> 528B row stride, conflict-free b128

typedef __attribute__((ext_vector_type(16))) __bf16 v16bf;
typedef __attribute__((ext_vector_type(8)))  float  v8f;

struct __attribute__((aligned(16))) U4 { unsigned int x, y, z, w; };

// ---- CDNA5 async global->LDS copy (16B per lane), tracked by ASYNCcnt ------
__device__ __forceinline__ void async_copy_b128(void* lds_ptr, const void* gptr) {
    unsigned lds_addr = (unsigned)(size_t)lds_ptr;          // low 32b = LDS offset
    unsigned long long gaddr = (unsigned long long)(size_t)gptr;
    asm volatile("global_load_async_to_lds_b128 %0, %1, off"
                 :: "v"(lds_addr), "v"(gaddr)
                 : "memory");
}
__device__ __forceinline__ void wait_async_zero() {
    asm volatile("s_wait_asynccnt 0" ::: "memory");
}

// ---------------------------------------------------------------------------
// zero accumulators (rowsum[2N], align accumulator)
// ---------------------------------------------------------------------------
__global__ void zero_kernel(float* __restrict__ rowsum, float* __restrict__ alignAcc) {
    int i = blockIdx.x * 256 + threadIdx.x;
    if (i < 2 * N_ROWS) rowsum[i] = 0.0f;
    if (i == 0) alignAcc[0] = 0.0f;
}

// ---------------------------------------------------------------------------
// prep: split fp32 -> bf16 hi/lo, row squared norms, align term
// ---------------------------------------------------------------------------
__global__ void prep_kernel(const float* __restrict__ v0,
                            const float* __restrict__ v1,
                            __bf16* __restrict__ hi,     // [2][N*D]
                            __bf16* __restrict__ lo,     // [2][N*D]
                            float*  __restrict__ sq,     // [2][N]
                            float*  __restrict__ alignAcc)
{
    const int row = blockIdx.x;
    const int k   = threadIdx.x;                // DIM == 256 == blockDim.x
    const size_t idx = (size_t)row * DIM + k;
    const size_t off1 = (size_t)N_ROWS * DIM;

    float x0 = v0[idx];
    float x1 = v1[idx];

    __bf16 h0 = (__bf16)x0;  __bf16 l0 = (__bf16)(x0 - (float)h0);
    __bf16 h1 = (__bf16)x1;  __bf16 l1 = (__bf16)(x1 - (float)h1);
    hi[idx] = h0;  hi[off1 + idx] = h1;
    lo[idx] = l0;  lo[off1 + idx] = l1;

    float s0 = x0 * x0;
    float s1 = x1 * x1;
    float dd = (x0 - x1) * (x0 - x1);

    #pragma unroll
    for (int m = 1; m < 32; m <<= 1) {
        s0 += __shfl_xor(s0, m, 32);
        s1 += __shfl_xor(s1, m, 32);
        dd += __shfl_xor(dd, m, 32);
    }
    __shared__ float red[3][8];
    const int wave = threadIdx.x >> 5;
    const int lane = threadIdx.x & 31;
    if (lane == 0) { red[0][wave] = s0; red[1][wave] = s1; red[2][wave] = dd; }
    __syncthreads();
    if (threadIdx.x == 0) {
        float t0 = 0.f, t1 = 0.f, td = 0.f;
        #pragma unroll
        for (int w = 0; w < 8; ++w) { t0 += red[0][w]; t1 += red[1][w]; td += red[2][w]; }
        sq[row]          = t0;
        sq[N_ROWS + row] = t1;
        atomicAdd(alignAcc, sqrtf(td));
    }
}

// ---------------------------------------------------------------------------
// fused Gram + exp row-sum kernel.
// grid = (N/BM, N/BN, 2 views), 256 threads = 8 waves.
// Each block: 64x128 Gram tile, full K=256.
// Wave w: row strip (w&3), 4 column sub-tiles ((w>>2)*4 + 0..3).
// Per k-step per wave: 1 A hi/lo fragment feeds 12 v_wmma.
// ---------------------------------------------------------------------------
__global__ __launch_bounds__(256)
void gram_entropy_kernel(const __bf16* __restrict__ hiBase,
                         const __bf16* __restrict__ loBase,
                         const float*  __restrict__ sqBase,
                         float*        __restrict__ rowsumBase)
{
    __shared__ __bf16 Ahi[BM][LDS_STRIDE] __attribute__((aligned(16)));
    __shared__ __bf16 Alo[BM][LDS_STRIDE] __attribute__((aligned(16)));
    __shared__ __bf16 Bhi[BN][LDS_STRIDE] __attribute__((aligned(16)));
    __shared__ __bf16 Blo[BN][LDS_STRIDE] __attribute__((aligned(16)));
    __shared__ float sqA[BM];
    __shared__ float sqB[BN];

    const int view = blockIdx.z;
    const __bf16* hi = hiBase + (size_t)view * N_ROWS * DIM;
    const __bf16* lo = loBase + (size_t)view * N_ROWS * DIM;
    const float*  sq = sqBase + (size_t)view * N_ROWS;
    float* rowsum    = rowsumBase + (size_t)view * N_ROWS;

    const int bm0 = blockIdx.x * BM;
    const int bn0 = blockIdx.y * BN;

    // ---- async stage A block (64 rows) hi/lo: 16B chunks per lane ----------
    for (int idx = threadIdx.x; idx < BM * (DIM / 8); idx += 256) {
        const int r = idx >> 5;           // 32 chunks per 256-elem row
        const int c = (idx & 31) * 8;
        async_copy_b128(&Ahi[r][c], &hi[(size_t)(bm0 + r) * DIM + c]);
        async_copy_b128(&Alo[r][c], &lo[(size_t)(bm0 + r) * DIM + c]);
    }
    // ---- async stage B block (128 rows) hi/lo ------------------------------
    for (int idx = threadIdx.x; idx < BN * (DIM / 8); idx += 256) {
        const int r = idx >> 5;
        const int c = (idx & 31) * 8;
        async_copy_b128(&Bhi[r][c], &hi[(size_t)(bn0 + r) * DIM + c]);
        async_copy_b128(&Blo[r][c], &lo[(size_t)(bn0 + r) * DIM + c]);
    }
    if (threadIdx.x < BM)            sqA[threadIdx.x]      = sq[bm0 + threadIdx.x];
    else if (threadIdx.x < BM + BN)  sqB[threadIdx.x - BM] = sq[bn0 + threadIdx.x - BM];
    wait_async_zero();      // this wave's async LDS writes are visible
    __syncthreads();        // all waves' writes visible

    const int wave = threadIdx.x >> 5;
    const int lane = threadIdx.x & 31;
    const int msub = wave & 3;            // 4 row strips of 16
    const int ng   = (wave >> 2) * 4;     // 4 column sub-tiles per wave

    // A-frag addressing (16-bit A 16x32 layout, ISA 7.12.2)
    const int lrow   = msub * 16 + (lane & 15);
    const int khalfA = (lane < 16) ? 0 : 8;
    // B-frag addressing: same K order as A (two 8-K chunks at khalfB and 16+khalfB)
    const int khalfB = (lane < 16) ? 0 : 8;
    int crow[4];
    #pragma unroll
    for (int t = 0; t < 4; ++t) crow[t] = (ng + t) * 16 + (lane & 15);

    v8f c[4];
    #pragma unroll
    for (int t = 0; t < 4; ++t) c[t] = (v8f){};

    // unroll capped at 2: ~one k-step of ds-load prefetch, VGPRs stay < 256
    #pragma unroll 2
    for (int kb = 0; kb < DIM; kb += 32) {
        v16bf ahi, alo;
        U4* p;
        p = (U4*)&ahi;
        p[0] = *(const U4*)&Ahi[lrow][kb + khalfA];
        p[1] = *(const U4*)&Ahi[lrow][kb + 16 + khalfA];
        p = (U4*)&alo;
        p[0] = *(const U4*)&Alo[lrow][kb + khalfA];
        p[1] = *(const U4*)&Alo[lrow][kb + 16 + khalfA];

        #pragma unroll
        for (int t = 0; t < 4; ++t) {
            v16bf bh, bl;
            p = (U4*)&bh;
            p[0] = *(const U4*)&Bhi[crow[t]][kb + khalfB];
            p[1] = *(const U4*)&Bhi[crow[t]][kb + khalfB + 16];
            p = (U4*)&bl;
            p[0] = *(const U4*)&Blo[crow[t]][kb + khalfB];
            p[1] = *(const U4*)&Blo[crow[t]][kb + khalfB + 16];

            // bf16x3: hi*hi + hi*lo + lo*hi, fp32 accumulate
            c[t] = __builtin_amdgcn_wmma_f32_16x16x32_bf16(false, ahi, false, bh, (short)0, c[t], false, false);
            c[t] = __builtin_amdgcn_wmma_f32_16x16x32_bf16(false, ahi, false, bl, (short)0, c[t], false, false);
            c[t] = __builtin_amdgcn_wmma_f32_16x16x32_bf16(false, alo, false, bh, (short)0, c[t], false, false);
        }
    }

    // ---- fused epilogue: d = max(sqrt(max(sq_i+sq_j-2g,0)), eps); sum exp(-d)
    const int hsel   = (lane < 16) ? 0 : 8;        // C/D: M = r + 8*(lane>=16)
    const int rowoff = bm0 + msub * 16 + hsel;
    float sqc[4];
    int   col[4];
    #pragma unroll
    for (int t = 0; t < 4; ++t) { sqc[t] = sqB[crow[t]]; col[t] = bn0 + crow[t]; }

    float acc[8];
    #pragma unroll
    for (int r = 0; r < 8; ++r) {
        const int row   = rowoff + r;
        const float sqr = sqA[msub * 16 + hsel + r];
        float tsum = 0.0f;
        #pragma unroll
        for (int t = 0; t < 4; ++t) {
            float d2 = fmaxf(sqr + sqc[t] - 2.0f * c[t][r], 0.0f);
            float d  = fmaxf(sqrtf(d2), EPS_F);
            tsum += (row == col[t]) ? 0.0f : __expf(-d);
        }
        acc[r] = tsum;
    }

    // reduce across the 16 lanes sharing the same rows (stay within half-wave)
    #pragma unroll
    for (int r = 0; r < 8; ++r) {
        #pragma unroll
        for (int m = 1; m < 16; m <<= 1)
            acc[r] += __shfl_xor(acc[r], m, 32);
    }
    const int r = lane & 15;
    if (r < 8)
        atomicAdd(&rowsum[rowoff + r], acc[r]);
}

// ---------------------------------------------------------------------------
// final: loss = align/N + [ sum(log rowsum) / (2N) - log(N-1) ]
// ---------------------------------------------------------------------------
__global__ void final_kernel(const float* __restrict__ rowsum,
                             const float* __restrict__ alignAcc,
                             float* __restrict__ out)
{
    float s = 0.0f;
    for (int i = threadIdx.x; i < 2 * N_ROWS; i += 256)
        s += logf(rowsum[i]);
    #pragma unroll
    for (int m = 1; m < 32; m <<= 1)
        s += __shfl_xor(s, m, 32);
    __shared__ float red[8];
    const int wave = threadIdx.x >> 5;
    const int lane = threadIdx.x & 31;
    if (lane == 0) red[wave] = s;
    __syncthreads();
    if (threadIdx.x == 0) {
        float tot = 0.0f;
        #pragma unroll
        for (int w = 0; w < 8; ++w) tot += red[w];
        const float entropy = tot / (2.0f * (float)N_ROWS) - logf((float)(N_ROWS - 1));
        out[0] = alignAcc[0] / (float)N_ROWS + entropy;
    }
}

// ---------------------------------------------------------------------------
extern "C" void kernel_launch(void* const* d_in, const int* in_sizes, int n_in,
                              void* d_out, int out_size, void* d_ws, size_t ws_size,
                              hipStream_t stream)
{
    (void)in_sizes; (void)n_in; (void)out_size; (void)ws_size;
    const float* v0 = (const float*)d_in[0];
    const float* v1 = (const float*)d_in[1];
    float* out = (float*)d_out;

    // workspace layout:
    //   hi  : 2*N*D bf16   (8 MB)
    //   lo  : 2*N*D bf16   (8 MB)
    //   sq  : 2*N f32
    //   rowsum : 2*N f32
    //   alignAcc : 1 f32
    __bf16* hi = (__bf16*)d_ws;
    __bf16* lo = hi + (size_t)2 * N_ROWS * DIM;
    float*  sqp      = (float*)(lo + (size_t)2 * N_ROWS * DIM);
    float*  rowsum   = sqp + 2 * N_ROWS;
    float*  alignAcc = rowsum + 2 * N_ROWS;

    zero_kernel<<<dim3((2 * N_ROWS + 255) / 256), dim3(256), 0, stream>>>(rowsum, alignAcc);
    prep_kernel<<<dim3(N_ROWS), dim3(256), 0, stream>>>(v0, v1, hi, lo, sqp, alignAcc);
    gram_entropy_kernel<<<dim3(N_ROWS / BM, N_ROWS / BN, 2), dim3(256), 0, stream>>>(hi, lo, sqp, rowsum);
    final_kernel<<<dim3(1), dim3(256), 0, stream>>>(rowsum, alignAcc, out);
}
